// FuCont_PSP_PC_11733850652677
// MI455X (gfx1250) — matrix operand order repaired and checked by reading, not compiled, not run on hardware
//
#include <hip/hip_runtime.h>
#include <math.h>

typedef _Float16 v16h __attribute__((ext_vector_type(16)));
typedef float    v8f  __attribute__((ext_vector_type(8)));

#define EPSF 1e-10f

// Problem constants (from setup_inputs)
#define BB   4
#define CC   512
#define CQ   128
#define HH   128
#define WW   128
#define HWN  16384     // H*W
#define NBIN 4
#define PH   32        // patch h
#define PW   32        // patch w
#define NP   64        // B*nbin*nbin patches

#if defined(__gfx1250__) && __has_builtin(__builtin_amdgcn_tensor_load_to_lds)
#define USE_TDM 1
#else
#define USE_TDM 0
#endif

// ---------------------------------------------------------------------------
// WMMA GEMM: C[M,N] = A[M,K] * B(K,N) (+bias[M]) (+gamma*acc + resid)
// A row-major (lda, K contiguous).
// BL==0: B element (k,n) at Bm[k*ldb + n]   (K-major rows, N contiguous)
// BL==1: B element (k,n) at Bm[n*ldb + k]   (transposed: K contiguous)
// Block tile 64(M) x 64(N), K step 32.
// 8 waves = 4(M) x 2(N); each wave: 16x32 -> 2 WMMAs sharing one A fragment.
// ---------------------------------------------------------------------------
#define APAD 34
#define BPAD 34

template<int BL, bool BIAS, bool RES>
__global__ __launch_bounds__(256)
void gemm_wmma_kernel(const float* __restrict__ A, int lda, long batchA,
                      const float* __restrict__ Bm, int ldb, long batchB,
                      float* __restrict__ Cm, int ldc, long batchC,
                      const float* __restrict__ bias,
                      const float* __restrict__ resid,
                      const float* __restrict__ gamma,
                      int M, int N, int K)
{
    __shared__ _Float16 As[64 * APAD];   // [m][k]
    __shared__ _Float16 Bs[64 * BPAD];   // [n][k] (transposed)

    const int t    = threadIdx.x;
    const int lane = t & 31;
    const int wid  = t >> 5;
    const int wm   = wid >> 1;   // 0..3  (M sub-tile)
    const int wn   = wid & 1;    // 0..1  (N sub-tile, 32 cols each)
    const int z    = blockIdx.z;
    const int m0   = blockIdx.y * 64;
    const int n0   = blockIdx.x * 64;

    A  += (long)z * batchA;
    Bm += (long)z * batchB;
    Cm += (long)z * batchC;
    if (RES) resid += (long)z * batchC;

    v8f acc0 = {0.f, 0.f, 0.f, 0.f, 0.f, 0.f, 0.f, 0.f};
    v8f acc1 = {0.f, 0.f, 0.f, 0.f, 0.f, 0.f, 0.f, 0.f};

    // A tile loader: row m = t>>2, 8 contiguous k at (t&3)*8
    const int am = t >> 2;
    const int ak = (t & 3) * 8;

    const int half = lane >> 4;          // 0 or 1
    const int l16  = lane & 15;
    const int mrow = wm * 16 + l16;      // A fragment row in tile
    const int nc0  = wn * 32 + l16;      // B fragment col, sub-tile 0
    const int nc1  = nc0 + 16;           // sub-tile 1

    const int nK = K / 32;
    for (int kk = 0; kk < nK; ++kk) {
        const int k0 = kk * 32;
        // ---- stage A (f32 -> f16), [m][k] with pad
        {
            const float* ap = A + (long)(m0 + am) * lda + (k0 + ak);
            _Float16* as = &As[am * APAD + ak];
            #pragma unroll
            for (int i = 0; i < 8; ++i) as[i] = (_Float16)ap[i];
            __builtin_prefetch(ap + 32, 0, 3);        // next k-tile
        }
        // ---- stage B transposed into [n][k] with pad
        if (BL == 0) {
            const int bk = t >> 3;            // 0..31
            const int bn = (t & 7) * 8;       // 0..56
            const float* bp = Bm + (long)(k0 + bk) * ldb + (n0 + bn);
            #pragma unroll
            for (int i = 0; i < 8; ++i) Bs[(bn + i) * BPAD + bk] = (_Float16)bp[i];
            __builtin_prefetch(bp + 32L * ldb, 0, 3); // next k-tile
        } else {
            const int bn = t >> 2;            // 0..63
            const int bk = (t & 3) * 8;       // 0..24
            const float* bp = Bm + (long)(n0 + bn) * ldb + (k0 + bk);
            #pragma unroll
            for (int i = 0; i < 8; ++i) Bs[bn * BPAD + bk + i] = (_Float16)bp[i];
            __builtin_prefetch(bp + 32, 0, 3);        // next k-tile
        }
        __syncthreads();

        // ---- build fragments per documented gfx1250 16-bit layouts
        v16h af, bf0, bf1;
        #pragma unroll
        for (int j = 0; j < 8; ++j) {
            // A 16x32: lanes 0-15 K=0..7,16..23 ; lanes 16-31 K=8..15,24..31
            const int kp = ((j < 4) ? 0 : 16) + 8 * half + 2 * (j & 3);
            af[2 * j]     = As[mrow * APAD + kp];
            af[2 * j + 1] = As[mrow * APAD + kp + 1];
        }
        #pragma unroll
        for (int e = 0; e < 16; ++e) {
            // B 32x16: lane = N, elements K = (e<8 ? 0 : 16) + 8*half + (e&7)
            bf0[e] = Bs[nc0 * BPAD + ((e < 8) ? 0 : 16) + 8 * half + (e & 7)];
            bf1[e] = Bs[nc1 * BPAD + ((e < 8) ? 0 : 16) + 8 * half + (e & 7)];
        }
        acc0 = __builtin_amdgcn_wmma_f32_16x16x32_f16(
                   false, af, false, bf0, (short)0, acc0, false, false);
        acc1 = __builtin_amdgcn_wmma_f32_16x16x32_f16(
                   false, af, false, bf1, (short)0, acc1, false, false);
        __syncthreads();
    }

    // ---- epilogue: C layout lane=N, vgpr r -> M = r + 8*half
    #pragma unroll
    for (int s = 0; s < 2; ++s) {
        const int col = n0 + wn * 32 + s * 16 + l16;
        #pragma unroll
        for (int r = 0; r < 8; ++r) {
            const int row = m0 + wm * 16 + r + 8 * half;
            float v = (s == 0) ? acc0[r] : acc1[r];
            if (BIAS) v += bias[row];
            const long off = (long)row * ldc + col;
            if (RES) v = gamma[0] * v + resid[off];
            Cm[off] = v;
        }
    }
}

// ---------------------------------------------------------------------------
// Channel L2 normalization (over Cc channels per spatial position), in place.
// ---------------------------------------------------------------------------
__global__ __launch_bounds__(256)
void chan_norm_kernel(float* __restrict__ q, int Cc, int total /* B*HW */)
{
    int idx = blockIdx.x * blockDim.x + threadIdx.x;
    if (idx >= total) return;
    int b = idx / HWN, n = idx % HWN;
    float* base = q + (long)b * Cc * HWN + n;
    float s = 0.f;
    for (int c = 0; c < Cc; ++c) { float v = base[(long)c * HWN]; s += v * v; }
    float r = 1.f / (sqrtf(s) + EPSF);
    for (int c = 0; c < Cc; ++c) base[(long)c * HWN] *= r;
}

// ---------------------------------------------------------------------------
// Criss-cross energies. ISH: fixed w, energy_H[h][g]; else fixed h, energy_W[w][g].
// ATT layout: [p][h][w][64]  (first 32 = H part, last 32 = W part)
// LDS tiles are staged by the Tensor Data Mover when available:
//   ISH : 3D tile (1, 32 h-lines stride W, 128 c-planes stride HW)
//   !ISH: 2D tile (32 contiguous w, 128 c-lines stride HW)
// Both land compacted as sX[c*32 + s].
// ---------------------------------------------------------------------------
template<bool ISH>
__global__ __launch_bounds__(256)
void cc_energy_kernel(const float* __restrict__ Qn, const float* __restrict__ Kn,
                      float* __restrict__ ATT)
{
    __shared__ float sQ[CQ * 32];
    __shared__ float sK[CQ * 32];
    const int p = blockIdx.x;
    const int fx = blockIdx.y;               // w (ISH) or h (!ISH)
    const int b = p >> 4, ib = (p >> 2) & 3, jb = p & 3;
    const int H0 = ib * PH, W0 = jb * PW;
    const long base = (long)b * CQ * HWN;
    const int t = threadIdx.x;
    const long elem0 = base + (ISH ? ((long)H0 * WW + (W0 + fx))
                                   : ((long)(H0 + fx) * WW + W0));

#if USE_TDM
    if (t < 32) {   // wave 0 drives the TDM
        typedef unsigned int v4u __attribute__((ext_vector_type(4)));
        typedef int          v8i __attribute__((ext_vector_type(8)));
        typedef int          v4i __attribute__((ext_vector_type(4)));
        typedef __attribute__((address_space(3))) char lds_char_t;

        const unsigned ldsQ = (unsigned)(unsigned long long)(lds_char_t*)sQ;
        const unsigned ldsK = (unsigned)(unsigned long long)(lds_char_t*)sK;
        const unsigned long long gq = (unsigned long long)(Qn + elem0);
        const unsigned long long gk = (unsigned long long)(Kn + elem0);

        const unsigned tile0 = ISH ? 1u   : 32u;
        const unsigned tile1 = ISH ? 32u  : 128u;
        const unsigned tile2 = ISH ? 128u : 0u;
        const unsigned long long s0 = ISH ? (unsigned long long)WW
                                          : (unsigned long long)HWN;
        const unsigned long long s1 = ISH ? (unsigned long long)HWN : 0ull;

        // D# group1: data_size=4B, huge tensor dims (no OOB clipping),
        // tile dims + 48-bit strides (element units).
        v8i g1 = { (int)(2u << 16),
                   (int)0u,                                  // [63:48]=dim0 lo16 = 0
                   (int)0x4000u,                             // dim0 hi16 (=2^30), dim1 lo16=0
                   (int)(0x4000u | (tile0 << 16)),           // dim1 hi16, tile_dim0
                   (int)(tile1 | (tile2 << 16)),             // tile_dim1, tile_dim2
                   (int)(unsigned)s0,                        // dim0_stride lo32
                   (int)((unsigned)(s0 >> 32) | (((unsigned)s1 & 0xFFFFu) << 16)),
                   (int)(unsigned)(s1 >> 16) };              // dim1_stride hi32
        v4i g2 = { (int)0x40000000, 0, 0, 0 };               // tensor_dim2 huge
        v4i g3 = { 0, 0, 0, 0 };
        v8i g4 = { 0, 0, 0, 0, 0, 0, 0, 0 };                 // extra group (clang-23 6-arg form)

        v4u g0q = { 1u, ldsQ, (unsigned)gq,
                    (unsigned)(((gq >> 32) & 0x1FFFFFFull) | (2ull << 30)) };
        v4u g0k = { 1u, ldsK, (unsigned)gk,
                    (unsigned)(((gk >> 32) & 0x1FFFFFFull) | (2ull << 30)) };

        __builtin_amdgcn_tensor_load_to_lds(g0q, g1, g2, g3, g4, 0);
        __builtin_amdgcn_tensor_load_to_lds(g0k, g1, g2, g3, g4, 0);
        __builtin_amdgcn_s_wait_tensorcnt(0);
    }
    __syncthreads();
#else
    #pragma unroll
    for (int i = 0; i < 16; ++i) {
        int idx = t + 256 * i;               // idx = c*32 + s
        int c = idx >> 5, s = idx & 31;
        long g = ISH ? (base + (long)c * HWN + (H0 + s) * WW + (W0 + fx))
                     : (base + (long)c * HWN + (H0 + fx) * WW + (W0 + s));
        sQ[idx] = Qn[g];
        sK[idx] = Kn[g];
    }
    __syncthreads();
#endif

    const int row = t >> 3;                  // h (ISH) or w (!ISH)
    const int g0i = (t & 7) * 4;
    float a0 = 0, a1 = 0, a2 = 0, a3 = 0;
    for (int c = 0; c < CQ; ++c) {
        float qv = sQ[c * 32 + row];
        const float* kk = &sK[c * 32 + g0i];
        a0 += qv * kk[0]; a1 += qv * kk[1]; a2 += qv * kk[2]; a3 += qv * kk[3];
    }
    long o = ISH ? ((((long)p * PH + row) * PW + fx) * 64 + g0i)
                 : ((((long)p * PH + fx) * PW + row) * 64 + 32 + g0i);
    ATT[o + 0] = a0; ATT[o + 1] = a1; ATT[o + 2] = a2; ATT[o + 3] = a3;
}

// ---------------------------------------------------------------------------
// 64-wide softmax per (p,h,w), in place on ATT.
// ---------------------------------------------------------------------------
__global__ __launch_bounds__(256)
void softmax64_kernel(float* __restrict__ ATT, int total)
{
    int idx = blockIdx.x * blockDim.x + threadIdx.x;
    if (idx >= total) return;
    float* p = ATT + (long)idx * 64;
    float v[64];
    float m = -3.0e38f;
    #pragma unroll
    for (int i = 0; i < 64; ++i) { v[i] = p[i]; m = fmaxf(m, v[i]); }
    float s = 0.f;
    #pragma unroll
    for (int i = 0; i < 64; ++i) { v[i] = __expf(v[i] - m); s += v[i]; }
    float r = 1.f / s;
    #pragma unroll
    for (int i = 0; i < 64; ++i) p[i] = v[i] * r;
}

// ---------------------------------------------------------------------------
// Apply: out[c,h,w] = sum_g v[c,g,w]*attH[h,w,g] + sum_g v[c,h,g]*attW[h,w,g]
// One block per (p, h).
// ---------------------------------------------------------------------------
__global__ __launch_bounds__(256)
void cc_apply_kernel(const float* __restrict__ V, const float* __restrict__ ATT,
                     float* __restrict__ dst)
{
    __shared__ float sAH[32 * 33];
    __shared__ float sAW[32 * 33];
    const int p = blockIdx.x, hh = blockIdx.y;
    const int b = p >> 4, ib = (p >> 2) & 3, jb = p & 3;
    const int H0 = ib * PH, W0 = jb * PW;
    const int t = threadIdx.x;
    #pragma unroll
    for (int i = 0; i < 4; ++i) {
        int idx = t + 256 * i;               // idx = w*32 + g
        int w = idx >> 5, g = idx & 31;
        const float* ap = ATT + (((long)p * PH + hh) * PW + w) * 64;
        sAH[w * 33 + g] = ap[g];
        sAW[w * 33 + g] = ap[32 + g];
    }
    __syncthreads();
    const int w  = t & 31;
    const int cs = t >> 5;                   // 0..7
    const long vb = (long)b * CC * HWN;
    for (int c = cs; c < CC; c += 8) {
        const float* vc   = V + vb + (long)c * HWN;
        const float* vcol = vc + (long)H0 * WW + (W0 + w);   // + g*WW
        const float* vrow = vc + (long)(H0 + hh) * WW + W0;  // + g
        float accH = 0.f, accW = 0.f;
        #pragma unroll 8
        for (int g = 0; g < 32; ++g) {
            accH += vcol[g * WW] * sAH[w * 33 + g];
            accW += vrow[g]      * sAW[w * 33 + g];
        }
        dst[vb + (long)c * HWN + (H0 + hh) * WW + (W0 + w)] = accH + accW;
    }
}

// ---------------------------------------------------------------------------
// CAM row norms: R[b*C+i] = sqrt(sum_n F^2) + EPS
// ---------------------------------------------------------------------------
__global__ __launch_bounds__(256)
void rownorm_kernel(const float* __restrict__ F, float* __restrict__ R)
{
    __shared__ float red[256];
    const long row = blockIdx.x;
    const float* p = F + row * (long)HWN;
    float s = 0.f;
    for (int n = threadIdx.x; n < HWN; n += 256) { float v = p[n]; s += v * v; }
    red[threadIdx.x] = s; __syncthreads();
    for (int o = 128; o > 0; o >>= 1) {
        if (threadIdx.x < o) red[threadIdx.x] += red[threadIdx.x + o];
        __syncthreads();
    }
    if (threadIdx.x == 0) R[row] = sqrtf(red[0]) + EPSF;
}

// ---------------------------------------------------------------------------
// CAM softmax: att[i][j] = softmax_j( G[i][j] / (R_i * R_j) ), in place on G.
// One block per row (b*C+i), 512 entries.
// ---------------------------------------------------------------------------
__global__ __launch_bounds__(256)
void cam_softmax_kernel(float* __restrict__ G, const float* __restrict__ R)
{
    __shared__ float red[256];
    const long row = blockIdx.x;
    const int b = (int)(blockIdx.x >> 9);
    const float ri = R[row];
    float* g = G + row * (long)CC;
    const float* Rb = R + (long)b * CC;
    const int t = threadIdx.x;

    float e0 = g[t]       / (ri * Rb[t]);
    float e1 = g[t + 256] / (ri * Rb[t + 256]);

    red[t] = fmaxf(e0, e1); __syncthreads();
    for (int o = 128; o > 0; o >>= 1) {
        if (t < o) red[t] = fmaxf(red[t], red[t + o]);
        __syncthreads();
    }
    const float m = red[0]; __syncthreads();

    e0 = __expf(e0 - m); e1 = __expf(e1 - m);
    red[t] = e0 + e1; __syncthreads();
    for (int o = 128; o > 0; o >>= 1) {
        if (t < o) red[t] += red[t + o];
        __syncthreads();
    }
    const float r = 1.f / red[0];
    g[t] = e0 * r; g[t + 256] = e1 * r;
}

// ---------------------------------------------------------------------------
// Host-side launch sequence
// ---------------------------------------------------------------------------
extern "C" void kernel_launch(void* const* d_in, const int* in_sizes, int n_in,
                              void* d_out, int out_size, void* d_ws, size_t ws_size,
                              hipStream_t stream)
{
    (void)in_sizes; (void)n_in; (void)out_size; (void)ws_size;

    const float* x_in  = (const float*)d_in[0];
    const float* w_q   = (const float*)d_in[1];
    const float* b_q   = (const float*)d_in[2];
    const float* w_k   = (const float*)d_in[3];
    const float* b_k   = (const float*)d_in[4];
    const float* w_v   = (const float*)d_in[5];
    const float* b_v   = (const float*)d_in[6];
    const float* gamma = (const float*)d_in[7];
    float* OUT = (float*)d_out;

    float* ws = (float*)d_ws;
    float* X1 = ws;                                   // B*C*HW
    float* Qb = X1 + (long)BB * CC * HWN;             // B*Cq*HW
    float* Kb = Qb + (long)BB * CQ * HWN;             // B*Cq*HW
    float* Vb = Kb + (long)BB * CQ * HWN;             // B*C*HW
    float* AT = Vb + (long)BB * CC * HWN;             // 64*32*32*64
    float* G  = AT + (long)NP * PH * PW * 64;         // B*C*C
    float* R  = G  + (long)BB * CC * CC;              // B*C

    const long xBatch = (long)CC * HWN;
    const long qBatch = (long)CQ * HWN;

    for (int it = 0; it < 2; ++it) {
        const float* src = (it == 0) ? x_in : OUT;
        float* dst = (it == 0) ? OUT : X1;

        // q = w_q @ x + b_q   (M=128, N=16384, K=512)
        gemm_wmma_kernel<0, true, false><<<dim3(HWN / 64, CQ / 64, BB), 256, 0, stream>>>(
            w_q, CC, 0L, src, HWN, xBatch, Qb, HWN, qBatch, b_q, nullptr, nullptr,
            CQ, HWN, CC);
        // k
        gemm_wmma_kernel<0, true, false><<<dim3(HWN / 64, CQ / 64, BB), 256, 0, stream>>>(
            w_k, CC, 0L, src, HWN, xBatch, Kb, HWN, qBatch, b_k, nullptr, nullptr,
            CQ, HWN, CC);
        // v  (M=512)
        gemm_wmma_kernel<0, true, false><<<dim3(HWN / 64, CC / 64, BB), 256, 0, stream>>>(
            w_v, CC, 0L, src, HWN, xBatch, Vb, HWN, xBatch, b_v, nullptr, nullptr,
            CC, HWN, CC);

        // normalize q,k over channels
        chan_norm_kernel<<<(BB * HWN + 255) / 256, 256, 0, stream>>>(Qb, CQ, BB * HWN);
        chan_norm_kernel<<<(BB * HWN + 255) / 256, 256, 0, stream>>>(Kb, CQ, BB * HWN);

        // criss-cross energies + softmax
        cc_energy_kernel<true ><<<dim3(NP, PW), 256, 0, stream>>>(Qb, Kb, AT);
        cc_energy_kernel<false><<<dim3(NP, PH), 256, 0, stream>>>(Qb, Kb, AT);
        softmax64_kernel<<<(NP * PH * PW + 255) / 256, 256, 0, stream>>>(AT, NP * PH * PW);

        // apply attention -> dst
        cc_apply_kernel<<<dim3(NP, PH), 256, 0, stream>>>(Vb, AT, dst);
    }

    // ---- CAM on X1 ----
    rownorm_kernel<<<BB * CC, 256, 0, stream>>>(X1, R);

    // G = F @ F^T  (M=N=512, K=16384), B transposed layout
    gemm_wmma_kernel<1, false, false><<<dim3(CC / 64, CC / 64, BB), 256, 0, stream>>>(
        X1, HWN, xBatch, X1, HWN, xBatch, G, CC, (long)CC * CC,
        nullptr, nullptr, nullptr, CC, CC, HWN);

    cam_softmax_kernel<<<BB * CC, 256, 0, stream>>>(G, R);

    // out = gamma * (att @ F) + F   (M=512, N=16384, K=512)
    gemm_wmma_kernel<0, false, true><<<dim3(HWN / 64, CC / 64, BB), 256, 0, stream>>>(
        G, CC, (long)CC * CC, X1, HWN, xBatch, OUT, HWN, xBatch,
        nullptr, X1, gamma, CC, HWN, CC);
}
